// PNALayer_30296699306204
// MI455X (gfx1250) — matrix operand order repaired and checked by reading, not compiled, not run on hardware
//
#include <hip/hip_runtime.h>
#include <hip/hip_bf16.h>

#define NN 50000
#define EE 800000
#define DD 96
#define PNA_LOG_AVG 2.8332133440562162f

typedef __bf16 bf16;
typedef __attribute__((ext_vector_type(16))) __bf16 v16bf;
typedef __attribute__((ext_vector_type(8)))  float  v8f;

// ---- monotonic float <-> uint encoding for atomicMax on floats ----
__device__ __forceinline__ unsigned encf(float f) {
  unsigned u = __float_as_uint(f);
  return (u >> 31) ? ~u : (u | 0x80000000u);
}
__device__ __forceinline__ float decf(unsigned u) {
  return (u >> 31) ? __uint_as_float(u & 0x7FFFFFFFu) : __uint_as_float(~u);
}

__device__ __forceinline__ v8f wmma_bf16(v16bf a, v16bf b, v8f c) {
  // 8 args: (neg_a, A, neg_b, B, c_mod, C, reuse_a, reuse_b)
  return __builtin_amdgcn_wmma_f32_16x16x32_bf16(false, a, false, b, (short)0, c,
                                                 false, false);
}

// A-fragment (16x32 bf16, row-major source) per ISA 7.12.2:
// lane 0-15: M=lane, K-lo half; lane 16-31: M=lane-16, K-hi half.
// VGPR v<4: K = 2v + 8*half (+pair); v>=4: K = 16 + 2(v-4) + 8*half (+pair)
// (contiguous runs -> compiler merges into ds_load_b128)
__device__ __forceinline__ v16bf load_afrag(const bf16* row, int half) {
  v16bf a;
#pragma unroll
  for (int v = 0; v < 8; ++v) {
    int k = ((v < 4) ? (2 * v) : (16 + 2 * (v - 4))) + 8 * half;
    a[2 * v]     = row[k];
    a[2 * v + 1] = row[k + 1];
  }
  return a;
}

// ---------------- setup kernels ----------------
__global__ void init_stats(float* sums, float* sumsq, unsigned* mxe, int* cnt) {
  int i = blockIdx.x * 256 + threadIdx.x;
  if (i < NN * DD) {
    sums[i] = 0.f;
    sumsq[i] = 0.f;
    mxe[i] = 0x007FFFFFu;   // enc(-inf)
  }
  if (i < NN) cnt[i] = 0;
}

__global__ void f2bf(const float* __restrict__ in, bf16* __restrict__ out, int n) {
  int i = blockIdx.x * 256 + threadIdx.x;
  if (i < n) out[i] = (bf16)in[i];
}

// Repack W[K x 96] (row-major, K = multiple of 32) into B-fragment-major order:
// out[((nt*KS + s)*32 + lane)*16 + j] = W[(32s + 8*(lane/16) + j + (j>=8 ? 8 : 0))*96 + nt*16 + lane%16]
__global__ void repack(const float* __restrict__ W, bf16* __restrict__ out, int K) {
  int o = blockIdx.x * 256 + threadIdx.x;
  if (o >= K * DD) return;
  int KS   = K / 32;
  int j    = o & 15;
  int lane = (o >> 4) & 31;
  int s    = (o >> 9) % KS;
  int nt   = (o >> 9) / KS;
  int k = 32 * s + ((lane >> 4) << 3) + j + ((j >> 3) << 3);
  int n = nt * 16 + (lane & 15);
  out[o] = (bf16)W[k * DD + n];
}

__global__ void degree(const int* __restrict__ dst, int* __restrict__ cnt) {
  int e = blockIdx.x * 256 + threadIdx.x;
  if (e < EE) atomicAdd(&cnt[dst[e]], 1);
}

// ---------------- edge kernel: pretrans GEMM + scatter stats ----------------
// block: 192 threads (6 waves), 64 edges (4 M-tiles). Each wave owns one 16-col
// N-tile and reuses every B fragment across the 4 M-tiles (4x less L2 weight
// traffic, 24 WMMAs per wave).
__global__ __launch_bounds__(192) void edge_kernel(
    const bf16* __restrict__ xb, const int* __restrict__ dst, const int* __restrict__ src,
    const bf16* __restrict__ wpk, const float* __restrict__ bpre,
    float* __restrict__ sums, float* __restrict__ sumsq, unsigned* __restrict__ mxe) {
  __shared__ bf16 Alds[64 * 200];  // 64 edges x 192 feats (+8 pad) = 25.6 KB
  __shared__ int  dstS[64];
  int tid = threadIdx.x;
  int e0 = blockIdx.x * 64;

  if (tid < 64) dstS[tid] = dst[e0 + tid];

  // cooperative gather: [x_dst | x_src] per edge, 32B vector copies
#pragma unroll
  for (int tt = tid; tt < 64 * 12; tt += 192) {
    int r = tt / 12, q = tt % 12;
    int e = e0 + r;
    int node = (q < 6) ? dst[e] : src[e];
    int c0 = (q % 6) * 16;
    const bf16* sp = xb + node * DD + c0;
    bf16* dp = &Alds[r * 200 + ((q < 6) ? 0 : DD) + c0];
    *(uint4*)dp       = *(const uint4*)sp;
    *(uint4*)(dp + 8) = *(const uint4*)(sp + 8);
  }
  __syncthreads();

  int lane = tid & 31, wv = tid >> 5;
  int half = lane >> 4, nloc = lane & 15;
  v8f acc0 = {0.f, 0.f, 0.f, 0.f, 0.f, 0.f, 0.f, 0.f};
  v8f acc1 = acc0, acc2 = acc0, acc3 = acc0;
  const bf16* wb = wpk + wv * 6 * 512;  // 6 K-steps for K=192
#pragma unroll
  for (int s = 0; s < 6; ++s) {
    v16bf b = *(const v16bf*)(wb + (s * 32 + lane) * 16);
    v16bf a0 = load_afrag(&Alds[(0 * 16 + nloc) * 200 + 32 * s], half);
    v16bf a1 = load_afrag(&Alds[(1 * 16 + nloc) * 200 + 32 * s], half);
    v16bf a2 = load_afrag(&Alds[(2 * 16 + nloc) * 200 + 32 * s], half);
    v16bf a3 = load_afrag(&Alds[(3 * 16 + nloc) * 200 + 32 * s], half);
    acc0 = wmma_bf16(a0, b, acc0);
    acc1 = wmma_bf16(a1, b, acc1);
    acc2 = wmma_bf16(a2, b, acc2);
    acc3 = wmma_bf16(a3, b, acc3);
  }

  int col = wv * 16 + nloc;
  float bp = bpre[col];
#pragma unroll
  for (int m = 0; m < 4; ++m) {
    v8f acc = (m == 0) ? acc0 : (m == 1) ? acc1 : (m == 2) ? acc2 : acc3;
#pragma unroll
    for (int r = 0; r < 8; ++r) {
      int er = m * 16 + r + 8 * half;     // edge row within tile
      int dnode = dstS[er];
      float v = acc[r] + bp;
      int o = dnode * DD + col;
      atomicAdd(&sums[o], v);
      atomicAdd(&sumsq[o], v * v);
      atomicMax(&mxe[o], encf(v));
    }
  }
}

// ---------------- node kernel: scalers + posttrans GEMM + mix GEMM ----------------
// block: 192 threads (6 waves), handles 16 nodes.
__global__ __launch_bounds__(192) void node_kernel(
    const bf16* __restrict__ xb, const int* __restrict__ cnt,
    const float* __restrict__ sums, const float* __restrict__ sumsq,
    const unsigned* __restrict__ mxe,
    const bf16* __restrict__ wpost_pk, const float* __restrict__ bpost,
    const bf16* __restrict__ wmix_pk, const float* __restrict__ bmix,
    float* __restrict__ out) {
  constexpr int AST = 1256;            // 1248 + 8 pad
  __shared__ bf16 Alds[16 * AST];      // 16 x 1248 PNA feature tile (~40 KB)
  __shared__ bf16 A2[16 * 104];        // 16 x 96 hpost tile (bf16)
  int tid = threadIdx.x;
  int n0 = blockIdx.x * 16;

  {  // build [x | aggrs | aggrs*amp | aggrs*att] row per node
    int r = tid / 12, q = tid % 12;
    int node = n0 + r;
    float c = (float)cnt[node];
    float cn = fmaxf(c, 1.0f);
    float l = logf(c + 1.0f);
    float la = l * (1.0f / PNA_LOG_AVG);
    float li = PNA_LOG_AVG / l;
    bf16* row = &Alds[r * AST];
#pragma unroll
    for (int dd = 0; dd < 8; ++dd) {
      int d = q * 8 + dd;
      int o = node * DD + d;
      float s = sums[o];
      float mx = (c > 0.f) ? decf(mxe[o]) : 0.f;
      float mean = s / cn;
      float var = fmaxf(sumsq[o] / cn - mean * mean, 0.f);
      row[d] = xb[o];
      row[96  + d] = (bf16)s;          row[192 + d] = (bf16)mx;
      row[288 + d] = (bf16)mean;       row[384 + d] = (bf16)var;
      row[480 + d] = (bf16)(s * la);   row[576 + d] = (bf16)(mx * la);
      row[672 + d] = (bf16)(mean * la);row[768 + d] = (bf16)(var * la);
      row[864 + d] = (bf16)(s * li);   row[960 + d] = (bf16)(mx * li);
      row[1056+ d] = (bf16)(mean * li);row[1152+ d] = (bf16)(var * li);
    }
  }
  __syncthreads();

  int lane = tid & 31, wv = tid >> 5;
  int half = lane >> 4, nloc = lane & 15;
  int col = wv * 16 + nloc;

  // GEMM1: (16 x 1248) @ W_post -> hpost tile ; 39 K-steps
  v8f acc = {0.f, 0.f, 0.f, 0.f, 0.f, 0.f, 0.f, 0.f};
  const bf16* wb = wpost_pk + wv * 39 * 512;
  for (int s = 0; s < 39; ++s) {
    if (s + 1 < 39) __builtin_prefetch(wb + ((s + 1) * 32 + lane) * 16, 0, 1);
    v16bf a = load_afrag(&Alds[nloc * AST + 32 * s], half);
    v16bf b = *(const v16bf*)(wb + (s * 32 + lane) * 16);
    acc = wmma_bf16(a, b, acc);
  }
  float bpv = bpost[col];
#pragma unroll
  for (int r = 0; r < 8; ++r)
    A2[(r + 8 * half) * 104 + col] = (bf16)(acc[r] + bpv);
  __syncthreads();

  // GEMM2: hpost @ W_mix ; 3 K-steps ; LeakyReLU(0.01)
  v8f acc2 = {0.f, 0.f, 0.f, 0.f, 0.f, 0.f, 0.f, 0.f};
  const bf16* wb2 = wmix_pk + wv * 3 * 512;
#pragma unroll
  for (int s = 0; s < 3; ++s) {
    v16bf a = load_afrag(&A2[nloc * 104 + 32 * s], half);
    v16bf b = *(const v16bf*)(wb2 + (s * 32 + lane) * 16);
    acc2 = wmma_bf16(a, b, acc2);
  }
  float bm = bmix[col];
#pragma unroll
  for (int r = 0; r < 8; ++r) {
    float v = acc2[r] + bm;
    out[(n0 + r + 8 * half) * DD + col] = (v >= 0.f) ? v : 0.01f * v;
  }
}

// ---------------- host launcher ----------------
extern "C" void kernel_launch(void* const* d_in, const int* in_sizes, int n_in,
                              void* d_out, int out_size, void* d_ws, size_t ws_size,
                              hipStream_t stream) {
  const float* x     = (const float*)d_in[0];
  const int*   ei    = (const int*)d_in[1];
  const int*   srcI  = ei;        // edge_index[0]
  const int*   dstI  = ei + EE;   // edge_index[1]
  const float* Wpre  = (const float*)d_in[3];
  const float* bpre  = (const float*)d_in[4];
  const float* Wpost = (const float*)d_in[5];
  const float* bpost = (const float*)d_in[6];
  const float* Wmix  = (const float*)d_in[7];
  const float* bmix  = (const float*)d_in[8];
  float* out = (float*)d_out;
  (void)in_sizes; (void)n_in; (void)out_size; (void)ws_size;

  unsigned char* base = (unsigned char*)d_ws;
  size_t off = 0;
  auto take = [&](size_t bytes) -> void* {
    void* p = base + off;
    off += (bytes + 255) & ~(size_t)255;
    return p;
  };
  bf16*     xb     = (bf16*)take((size_t)NN * DD * 2);
  bf16*     wprep  = (bf16*)take((size_t)192 * DD * 2);
  bf16*     wpostp = (bf16*)take((size_t)1248 * DD * 2);
  bf16*     wmixp  = (bf16*)take((size_t)96 * DD * 2);
  int*      cnt    = (int*)take((size_t)NN * 4);
  float*    sums   = (float*)take((size_t)NN * DD * 4);
  float*    sumsq  = (float*)take((size_t)NN * DD * 4);
  unsigned* mxe    = (unsigned*)take((size_t)NN * DD * 4);

  init_stats<<<(NN * DD + 255) / 256, 256, 0, stream>>>(sums, sumsq, mxe, cnt);
  f2bf<<<(NN * DD + 255) / 256, 256, 0, stream>>>(x, xb, NN * DD);
  repack<<<(192 * DD + 255) / 256, 256, 0, stream>>>(Wpre, wprep, 192);
  repack<<<(1248 * DD + 255) / 256, 256, 0, stream>>>(Wpost, wpostp, 1248);
  repack<<<(96 * DD + 255) / 256, 256, 0, stream>>>(Wmix, wmixp, 96);
  degree<<<(EE + 255) / 256, 256, 0, stream>>>(dstI, cnt);
  edge_kernel<<<EE / 64, 192, 0, stream>>>(xb, dstI, srcI, wprep, bpre,
                                           sums, sumsq, mxe);
  node_kernel<<<NN / 16, 192, 0, stream>>>(xb, cnt, sums, sumsq, mxe,
                                           wpostp, bpost, wmixp, bmix, out);
}
